// TripletMarginLoss_66924180407504
// MI455X (gfx1250) — compile-verified, matrix operand order checked by reading, not run on hardware
//
#include <hip/hip_runtime.h>
#include <hip/hip_bf16.h>

typedef __attribute__((ext_vector_type(16))) _Float16 v16h;
typedef __attribute__((ext_vector_type(8)))  _Float16 h8;   // 16-byte chunk
typedef __attribute__((ext_vector_type(8)))  float    v8f;

constexpr int NB  = 8192;   // batch
constexpr int GD  = 256;    // graph dim
constexpr int TD  = 384;    // text dim
constexpr int KT  = TD / 32;     // 12 WMMA k-steps per 16x16 sim tile
constexpr int CP  = 32;          // columns staged in LDS per chunk (double-buffered)
constexpr int LDSW = TD + 8;     // padded f16 row stride: 392*2=784B -> banks 0,4,..,60
constexpr int PANEL = CP * LDSW; // f16 elements per panel buffer

// ---------------------------------------------------------------- kernel 1
// t_n = l2norm(text) rows, stored f16 (range-safe: |x| <= 1 after normalize)
__global__ __launch_bounds__(128)
void tnorm_kernel(const float* __restrict__ t, _Float16* __restrict__ o) {
  const int row = blockIdx.x, tid = threadIdx.x;
  const float* r = t + (size_t)row * TD;
  float x0 = r[tid], x1 = r[tid + 128], x2 = r[tid + 256];
  float s = x0 * x0 + x1 * x1 + x2 * x2;
  #pragma unroll
  for (int off = 16; off; off >>= 1) s += __shfl_xor(s, off);
  __shared__ float red[4];
  __shared__ float sinv;
  if ((tid & 31) == 0) red[tid >> 5] = s;
  __syncthreads();
  if (tid == 0) {
    float tot = red[0] + red[1] + red[2] + red[3];
    sinv = 1.0f / fmaxf(sqrtf(tot), 1e-12f);
  }
  __syncthreads();
  const float inv = sinv;
  _Float16* w = o + (size_t)row * TD;
  w[tid]       = (_Float16)(x0 * inv);
  w[tid + 128] = (_Float16)(x1 * inv);
  w[tid + 256] = (_Float16)(x2 * inv);
}

// ---------------------------------------------------------------- kernel 2
// g = l2norm(graph @ W^T + b), f32. 16 rows per block, 384 threads (1 col each).
__global__ __launch_bounds__(384)
void gproj_kernel(const float* __restrict__ ge, const float* __restrict__ W,
                  const float* __restrict__ bias, float* __restrict__ g) {
  __shared__ float gsh[16 * GD];     // 16 KB
  __shared__ float ysh[16 * TD];     // 24 KB
  __shared__ float rinv[16];
  const int tid = threadIdx.x;
  const int rowbase = blockIdx.x * 16;
  for (int idx = tid; idx < 16 * GD; idx += 384)
    gsh[idx] = ge[(size_t)rowbase * GD + idx];
  __syncthreads();
  float acc[16];
  #pragma unroll
  for (int r = 0; r < 16; ++r) acc[r] = 0.0f;
  const float* wrow = W + (size_t)tid * GD;
  for (int k = 0; k < GD; ++k) {
    const float w = wrow[k];
    #pragma unroll
    for (int r = 0; r < 16; ++r) acc[r] += gsh[r * GD + k] * w;
  }
  const float bj = bias[tid];
  #pragma unroll
  for (int r = 0; r < 16; ++r) ysh[r * TD + tid] = acc[r] + bj;
  __syncthreads();
  if (tid < 16) {
    float s = 0.0f;
    for (int j = 0; j < TD; ++j) { float v = ysh[tid * TD + j]; s += v * v; }
    rinv[tid] = 1.0f / fmaxf(sqrtf(s), 1e-12f);
  }
  __syncthreads();
  #pragma unroll
  for (int r = 0; r < 16; ++r)
    g[(size_t)(rowbase + r) * TD + tid] = ysh[r * TD + tid] * rinv[r];
}

// ---------------------------------------------------------------- kernel 3
// sim = t_n @ t_n^T via v_wmma_f32_16x16x32_f16 with fused per-row
// argmax (diag masked to 0, included) / argmin (diag excluded).
// 4 waves x 32 rows = 128 rows per block. Two A panels (16x384 each) in VGPRs
// so each B tile read from LDS feeds TWO wmma ops (ds:wmma = 1:1).
// B panels are double-buffered in LDS and filled with ASYNC loads
// (global_load_async_to_lds_b128, ASYNCcnt) overlapped with compute.

__device__ __forceinline__ void async_ld16_to_lds(const _Float16* gsrc, uint32_t lds_byte) {
  // VGLOBAL async encoding, GV mode (SADDR=off): copies 16 B from global vaddr into LDS at byte address vdst
  asm volatile("global_load_async_to_lds_b128 %0, %1, off"
               :: "v"(lds_byte), "v"(gsrc) : "memory");
}

__device__ __forceinline__ void stage_panel_async(const _Float16* __restrict__ t,
                                                  int cbase, _Float16* dst, int tid) {
  // CP rows x (TD/8)=48 chunks of 16B; 128 threads -> 12 async loads each
  #pragma unroll 1
  for (int idx = tid; idx < CP * (TD / 8); idx += 128) {
    const int r = idx / (TD / 8), c = idx % (TD / 8);
    const uint32_t lds = (uint32_t)(uintptr_t)(dst + r * LDSW + c * 8);
    async_ld16_to_lds(t + (size_t)(cbase + r) * TD + c * 8, lds);
  }
}

__global__ __launch_bounds__(128)
void sim_argext_kernel(const _Float16* __restrict__ t,
                       int* __restrict__ pos, int* __restrict__ neg) {
  __shared__ _Float16 bsh[2 * PANEL];            // 50,176 B (double buffer)
  const int tid  = threadIdx.x;
  const int lane = tid & 31;
  const int wave = tid >> 5;
  const int hi   = lane >> 4;        // half-wave select (K striping)
  const int lm   = lane & 15;        // M (A) or N (B) within tile
  const int mbase = blockIdx.x * 128 + wave * 32;   // 32 rows per wave (2 tiles)

  // Two A panels: rows [mbase, mbase+16) and [mbase+16, mbase+32).
  // Per-lane layout: M = lm; elems 0..7 -> K = k0 + hi*8 + e,
  //                          elems 8..15 -> K = k0 + 16 + hi*8 + e.
  v16h apan[2][KT];
  #pragma unroll
  for (int tt = 0; tt < 2; ++tt) {
    const _Float16* ar = t + (size_t)(mbase + tt * 16 + lm) * TD;
    #pragma unroll
    for (int kb = 0; kb < KT; ++kb) {
      v16h a;
      *((h8*)&a)     = *(const h8*)(ar + kb * 32 + hi * 8);
      *((h8*)&a + 1) = *(const h8*)(ar + kb * 32 + 16 + hi * 8);
      apan[tt][kb] = a;
    }
  }

  float mxv[2][8], mnv[2][8];
  int   mxi[2][8], mni[2][8];
  #pragma unroll
  for (int tt = 0; tt < 2; ++tt)
    #pragma unroll
    for (int r = 0; r < 8; ++r) {
      mxv[tt][r] = -3.0e38f; mnv[tt][r] = 3.0e38f; mxi[tt][r] = 0; mni[tt][r] = 0;
    }

  constexpr int NCH = NB / CP;   // 256 chunks
  stage_panel_async(t, 0, bsh, tid);          // prologue: fill buffer 0

  for (int cc = 0; cc < NCH; ++cc) {
    asm volatile("s_wait_asynccnt 0" ::: "memory");   // my loads for buf[cc&1] done
    __syncthreads();                                   // everyone's done; prev reads done
    if (cc + 1 < NCH)                                  // prefetch next panel async
      stage_panel_async(t, (cc + 1) * CP, bsh + ((cc + 1) & 1) * PANEL, tid);

    const _Float16* buf = bsh + (cc & 1) * PANEL;
    const int cbase = cc * CP;
    #pragma unroll
    for (int sub = 0; sub < CP / 16; ++sub) {
      v8f c[2] = {{}, {}};
      const _Float16* bc = buf + (sub * 16 + lm) * LDSW;  // B: N = lm, same K striping
      #pragma unroll
      for (int kb = 0; kb < KT; ++kb) {
        v16h b;
        *((h8*)&b)     = *(const h8*)(bc + kb * 32 + hi * 8);
        *((h8*)&b + 1) = *(const h8*)(bc + kb * 32 + 16 + hi * 8);
        c[0] = __builtin_amdgcn_wmma_f32_16x16x32_f16(false, apan[0][kb], false, b,
                                                      (short)0, c[0], false, false);
        c[1] = __builtin_amdgcn_wmma_f32_16x16x32_f16(false, apan[1][kb], false, b,
                                                      (short)0, c[1], false, false);
      }
      const int n = cbase + sub * 16 + lm;
      #pragma unroll
      for (int tt = 0; tt < 2; ++tt) {
        #pragma unroll
        for (int r = 0; r < 8; ++r) {
          const int m = mbase + tt * 16 + r + 8 * hi;
          const float v  = c[tt][r];
          const float vm = (n == m) ? 0.0f : v;   // multiplicative diag mask (argmax)
          if (vm > mxv[tt][r]) { mxv[tt][r] = vm; mxi[tt][r] = n; }
          if (n != m && v < mnv[tt][r]) { mnv[tt][r] = v; mni[tt][r] = n; }
        }
      }
    }
  }

  // reduce across the 16 lanes of each half-wave (columns n%16 = lm)
  #pragma unroll
  for (int tt = 0; tt < 2; ++tt) {
    #pragma unroll
    for (int r = 0; r < 8; ++r) {
      float v = mxv[tt][r]; int i = mxi[tt][r];
      #pragma unroll
      for (int off = 1; off <= 8; off <<= 1) {
        float ov = __shfl_xor(v, off); int oi = __shfl_xor(i, off);
        if (ov > v || (ov == v && oi < i)) { v = ov; i = oi; }
      }
      mxi[tt][r] = i;
      float w = mnv[tt][r]; int j = mni[tt][r];
      #pragma unroll
      for (int off = 1; off <= 8; off <<= 1) {
        float ow = __shfl_xor(w, off); int oj = __shfl_xor(j, off);
        if (ow < w || (ow == w && oj < j)) { w = ow; j = oj; }
      }
      mni[tt][r] = j;
    }
  }
  if (lm == 0) {
    #pragma unroll
    for (int tt = 0; tt < 2; ++tt)
      #pragma unroll
      for (int r = 0; r < 8; ++r) {
        pos[mbase + tt * 16 + 8 * hi + r] = mxi[tt][r];
        neg[mbase + tt * 16 + 8 * hi + r] = mni[tt][r];
      }
  }
}

// ---------------------------------------------------------------- kernel 4/5
__global__ __launch_bounds__(256)
void loss_partial_kernel(const float* __restrict__ g, const int* __restrict__ pos,
                         const int* __restrict__ neg, float* __restrict__ part) {
  const int tid = threadIdx.x;
  const int i = blockIdx.x * 256 + tid;
  const float* a = g + (size_t)i * TD;
  const float* p = g + (size_t)pos[i] * TD;
  const float* q = g + (size_t)neg[i] * TD;
  float dp = 0.0f, dn = 0.0f;
  for (int d = 0; d < TD; ++d) {
    const float av = a[d];
    const float e1 = av - p[d]; dp += e1 * e1;
    const float e2 = av - q[d]; dn += e2 * e2;
  }
  float l = fmaxf(dp - dn + 0.5f, 0.0f);
  #pragma unroll
  for (int off = 16; off; off >>= 1) l += __shfl_xor(l, off);
  __shared__ float red[8];
  if ((tid & 31) == 0) red[tid >> 5] = l;
  __syncthreads();
  if (tid == 0) {
    float s = 0.0f;
    #pragma unroll
    for (int w = 0; w < 8; ++w) s += red[w];
    part[blockIdx.x] = s;
  }
}

__global__ void loss_final_kernel(const float* __restrict__ part, float* __restrict__ out) {
  float s = part[threadIdx.x];
  #pragma unroll
  for (int off = 16; off; off >>= 1) s += __shfl_xor(s, off);
  if (threadIdx.x == 0) out[0] = s * (1.0f / (float)NB);
}

// ---------------------------------------------------------------- launch
extern "C" void kernel_launch(void* const* d_in, const int* in_sizes, int n_in,
                              void* d_out, int out_size, void* d_ws, size_t ws_size,
                              hipStream_t stream) {
  const float* ge   = (const float*)d_in[0];   // [8192,256]
  const float* te   = (const float*)d_in[1];   // [8192,384]
  const float* W    = (const float*)d_in[2];   // [384,256]
  const float* bias = (const float*)d_in[3];   // [384]

  char* ws = (char*)d_ws;
  const size_t SZ_T = (size_t)NB * TD * sizeof(_Float16);      // 6,291,456
  const size_t SZ_G = (size_t)NB * TD * sizeof(float);         // 12,582,912
  _Float16* tF  = (_Float16*)ws;
  float*    g   = (float*)(ws + SZ_T);
  int*      pos = (int*)(ws + SZ_T + SZ_G);
  int*      neg = (int*)(ws + SZ_T + SZ_G + (size_t)NB * 4);
  float*    prt = (float*)(ws + SZ_T + SZ_G + (size_t)NB * 8);

  tnorm_kernel<<<NB, 128, 0, stream>>>(te, tF);
  gproj_kernel<<<NB / 16, 384, 0, stream>>>(ge, W, bias, g);
  sim_argext_kernel<<<NB / 128, 128, 0, stream>>>(tF, pos, neg);
  loss_partial_kernel<<<NB / 256, 256, 0, stream>>>(g, pos, neg, prt);
  loss_final_kernel<<<1, 32, 0, stream>>>(prt, (float*)d_out);
}
